// CNN_TPN_7559142441315
// MI455X (gfx1250) — matrix operand order repaired and checked by reading, not compiled, not run on hardware
//
#include <hip/hip_runtime.h>

typedef __attribute__((ext_vector_type(16))) _Float16 v16h;
typedef __attribute__((ext_vector_type(8)))  _Float16 v8h;
typedef __attribute__((ext_vector_type(8)))  float    v8f;
typedef __attribute__((ext_vector_type(2)))  float    v2f;

#define EPSF 2.220446049250313e-16f

// ---------------------------------------------------------------- utilities
__global__ void k_fill_zero(float* p, int n) {
  int i = blockIdx.x * blockDim.x + threadIdx.x;
  if (i < n) p[i] = 0.f;
}

// weights f32 [K][64] -> f16 swizzled [Kpad/32][64][32] so a lane's B fragment
// (16 consecutive k at fixed output channel) is 32 contiguous bytes.
__global__ void k_cvt_wsw(const float* w, _Float16* o, int K, int Kpad) {
  int i = blockIdx.x * blockDim.x + threadIdx.x;
  int n = Kpad * 64;
  if (i >= n) return;
  int kk   = i & 31;
  int t    = i >> 5;
  int c    = t & 63;
  int kb32 = t >> 6;
  int k = kb32 * 32 + kk;
  o[i] = (k < K) ? (_Float16)w[k * 64 + c] : (_Float16)0.f;
}

__global__ void k_f32_to_f16(const float* a, _Float16* o, int n) {
  int i = blockIdx.x * blockDim.x + threadIdx.x;
  if (i < n) o[i] = (_Float16)a[i];
}

// fused BN constants: y = sc*conv + bi  with sc=g*rsqrt(mv+1e-3), bi=(b-mm)*sc+be
__global__ void k_bnprep(const float* g, const float* be, const float* mm,
                         const float* mv, const float* b, float* sc, float* bi, int C) {
  int i = threadIdx.x;
  if (i >= C) return;
  float s = g[i] * rsqrtf(mv[i] + 1e-3f);
  sc[i] = s;
  bi[i] = (b[i] - mm[i]) * s + be[i];
}

// gather 25 images (global index base..base+24) from s (25 imgs) then q (150 imgs), f32->f16
__global__ void k_pack_input(const float* s, const float* q, _Float16* o, int base) {
  int i = blockIdx.x * blockDim.x + threadIdx.x;
  const int per = 84 * 84 * 3;
  if (i >= 25 * per) return;
  int li = i / per, off = i - li * per;
  int g = base + li;
  float v = (g < 25) ? s[g * per + off] : q[(g - 25) * per + off];
  o[i] = (_Float16)v;
}

// ------------------------------------------------- implicit-GEMM 3x3 conv (WMMA f16)
// grid: (ceil(H*W/16), Nimg), block: 128 (4 waves). Each wave does a 16x16 tile of
// [pixels x out-channels]; 64 out channels total. Patch (im2col rows) staged in LDS.
// wts is the swizzled layout from k_cvt_wsw.
__global__ __launch_bounds__(128)
void k_conv3x3_wmma(const _Float16* __restrict__ in, const _Float16* __restrict__ wts,
                    const float* __restrict__ scale, const float* __restrict__ bias,
                    _Float16* __restrict__ out, int H, int W, int Cin, int Kpad) {
  const int img  = blockIdx.y;
  const int tile = blockIdx.x;
  const int HW = H * W;
  const int K  = 9 * Cin;
  alignas(16) __shared__ _Float16 patch[16 * 576];

  const _Float16* inImg = in + (size_t)img * HW * Cin;
  for (int t = threadIdx.x; t < 16 * Kpad; t += 128) {
    int r = t / Kpad, k = t - r * Kpad;
    int p = tile * 16 + r;
    _Float16 v = (_Float16)0.f;
    if (p < HW && k < K) {
      int ky  = k / (3 * Cin);
      int rem = k - ky * 3 * Cin;
      int kx  = rem / Cin;
      int ci  = rem - kx * Cin;
      int py = p / W, px = p - py * W;
      int iy = py + ky - 1, ix = px + kx - 1;
      if (iy >= 0 && iy < H && ix >= 0 && ix < W)
        v = inImg[(iy * W + ix) * Cin + ci];
    }
    patch[r * Kpad + k] = v;
  }
  __syncthreads();

  const int lane = threadIdx.x & 31;
  const int wv   = threadIdx.x >> 5;
  const int co0  = wv * 16;
  const int h    = lane & 15;
  const int g2   = lane >> 4;

  // per-lane base pointers
  const _Float16* aprow = patch + h * Kpad + (g2 << 4);          // + kb, +8
  const _Float16* bbase = wts + (((size_t)(co0 + h)) << 5) + (g2 << 4);  // + kb32*2048

  v8f acc = {};
  for (int kb = 0; kb < Kpad; kb += 32) {
    // A fragment, contiguous k order (same as B): lane(h,g2) loads k = kb+16*g2+[0..15]
    v8h a0 = *(const v8h*)(aprow + kb);
    v8h a1 = *(const v8h*)(aprow + kb + 8);
    v16h a = __builtin_shufflevector(a0, a1, 0, 1, 2, 3, 4, 5, 6, 7,
                                     8, 9, 10, 11, 12, 13, 14, 15);
    // B 32x16 f16 layout: lane(h,g2) needs 16 consecutive swizzled halves
    const _Float16* bp = bbase + ((size_t)(kb >> 5) << 11);
    v16h b = *(const v16h*)bp;
    if (kb + 32 < Kpad)
      __builtin_prefetch(bp + 2048, 0, 3);  // next K-block weights -> global_prefetch
    acc = __builtin_amdgcn_wmma_f32_16x16x32_f16(false, a, false, b, (short)0, acc,
                                                 false, false);
  }

  const int n = co0 + h;
  const float sc = scale[n], bi = bias[n];
  _Float16* outImg = out + (size_t)img * HW * 64;
  if (tile * 16 + 16 <= HW) {
    // full tile: unconditional stores
#pragma unroll
    for (int r = 0; r < 8; ++r) {
      int p = tile * 16 + r + (g2 << 3);  // C/D layout: VGPR r -> M = r + 8*(lane>=16)
      float y = fmaxf(acc[r] * sc + bi, 0.f);
      outImg[p * 64 + n] = (_Float16)y;
    }
  } else {
#pragma unroll
    for (int r = 0; r < 8; ++r) {
      int p = tile * 16 + r + (g2 << 3);
      if (p < HW) {
        float y = fmaxf(acc[r] * sc + bi, 0.f);
        outImg[p * 64 + n] = (_Float16)y;
      }
    }
  }
}

// ------------------------------------------------- 2x2/stride2 maxpool, clipped windows
__global__ void k_maxpool2(const _Float16* in, _Float16* out16, float* out32,
                           int Nimg, int H, int W, int Ho, int Wo) {
  int idx = blockIdx.x * blockDim.x + threadIdx.x;
  int total = Nimg * Ho * Wo * 64;
  if (idx >= total) return;
  int c = idx & 63;
  int t = idx >> 6;
  int ox = t % Wo; t /= Wo;
  int oy = t % Ho; int img = t / Ho;
  const _Float16* ip = in + (size_t)img * H * W * 64;
  float m = -3.4e38f;
  for (int dy = 0; dy < 2; ++dy) {
    int iy = 2 * oy + dy; if (iy >= H) continue;
    for (int dx = 0; dx < 2; ++dx) {
      int ix = 2 * ox + dx; if (ix >= W) continue;
      float v = (float)ip[(iy * W + ix) * 64 + c];
      m = v > m ? v : m;
    }
  }
  if (out16) out16[idx] = (_Float16)m;
  if (out32) out32[idx] = m;
}

// --------------------------------- relation net tail: c2 conv (3x3x64->1) + pool + MLP
__global__ void k_relation_tail(const _Float16* r1p, const float* wc2,
                                const float* sc2, const float* bi2,
                                const float* w1, const float* b1,
                                const float* w2, const float* b2, float* sigma) {
  __shared__ float s9[9];
  int img = blockIdx.x, t = threadIdx.x;
  if (t < 9) {
    int oy = t / 3, ox = t - 3 * oy;
    const _Float16* ip = r1p + (size_t)img * 9 * 64;
    float acc = 0.f;
    for (int ky = 0; ky < 3; ++ky) {
      int iy = oy + ky - 1; if (iy < 0 || iy >= 3) continue;
      for (int kx = 0; kx < 3; ++kx) {
        int ix = ox + kx - 1; if (ix < 0 || ix >= 3) continue;
        const _Float16* px = ip + (iy * 3 + ix) * 64;
        const float* wk = wc2 + (ky * 3 + kx) * 64;
        for (int c = 0; c < 64; ++c) acc += (float)px[c] * wk[c];
      }
    }
    float x = acc * sc2[0] + bi2[0];
    s9[t] = x > 0.f ? x : 0.f;
  }
  __syncthreads();
  if (t == 0) {
    float p00 = fmaxf(fmaxf(s9[0], s9[1]), fmaxf(s9[3], s9[4]));
    float p01 = fmaxf(s9[2], s9[5]);
    float p10 = fmaxf(s9[6], s9[7]);
    float p11 = s9[8];
    float x4[4] = {p00, p01, p10, p11};
    float o = b2[0];
    for (int j = 0; j < 8; ++j) {
      float hh = b1[j];
      for (int i = 0; i < 4; ++i) hh += x4[i] * w1[i * 8 + j];
      hh = hh > 0.f ? hh : 0.f;
      o += hh * w2[j];
    }
    sigma[img] = 1.f / (1.f + expf(-o));
  }
}

// ------------------------------------------------------------- graph construction
__global__ void k_build_xn(const float* feat, const float* sigma, float* xnp) {
  int i = blockIdx.x * blockDim.x + threadIdx.x;
  if (i >= 175 * 1600) return;
  int r = i / 1600;
  xnp[i] = feat[i] / (sigma[r] + EPSF);  // row 175 of xnp stays zero (pre-filled)
}

// Gram G = xn @ xn^T over padded 176x1600, f32 WMMA 16x16x4. grid 31 x 128 threads.
__global__ __launch_bounds__(128)
void k_gram_wmma(const float* __restrict__ xnp, float* __restrict__ G) {
  int tile = blockIdx.x * 4 + (threadIdx.x >> 5);
  if (tile >= 121) return;  // wave-uniform exit
  int tm = tile / 11, tn = tile - 11 * tm;
  int lane = threadIdx.x & 31, h = lane & 15, g2 = lane >> 4;
  const float* arow = xnp + (size_t)(tm * 16 + h) * 1600 + 2 * g2;
  const float* brow = xnp + (size_t)(tn * 16 + h) * 1600 + 2 * g2;
  v8f acc = {};
#pragma unroll 4
  for (int kb = 0; kb < 1600; kb += 4) {
    // f32 A 16x4 layout: VGPR0 = K(2*g2), VGPR1 = K(2*g2+1) -> one b64 load
    v2f a = *(const v2f*)(arow + kb);
    v2f b = *(const v2f*)(brow + kb);
    acc = __builtin_amdgcn_wmma_f32_16x16x4_f32(false, a, false, b, (short)0, acc,
                                                false, false);
  }
#pragma unroll
  for (int r = 0; r < 8; ++r)
    G[(size_t)(tm * 16 + r + 8 * g2) * 176 + tn * 16 + h] = acc[r];
}

__global__ void k_wm(const float* G, float* Wm) {
  int idx = blockIdx.x * blockDim.x + threadIdx.x;
  if (idx >= 175 * 175) return;
  int i = idx / 175, j = idx - 175 * i;
  float d = G[i * 176 + i] + G[j * 176 + j] - 2.f * G[i * 176 + j];
  d = fmaxf(d, 0.f) * (1.f / 1600.f);
  Wm[idx] = expf(-0.5f * d);
}

// top-20 per row (stable: ties keep lower index), symmetric OR sparsify into W2
__global__ void k_topk_mask(const float* Wm, float* W2) {
  int i = threadIdx.x;
  if (i >= 175) return;
  float tv[20]; int ti[20];
  for (int t = 0; t < 20; ++t) { tv[t] = -1e30f; ti[t] = 0; }
  for (int j = 0; j < 175; ++j) {
    float v = Wm[i * 175 + j];
    if (v > tv[19]) {
      int t = 19;
      while (t > 0 && tv[t - 1] < v) { tv[t] = tv[t - 1]; ti[t] = ti[t - 1]; --t; }
      tv[t] = v; ti[t] = j;
    }
  }
  for (int t = 0; t < 20; ++t) {
    int j = ti[t];
    W2[i * 175 + j] = Wm[i * 175 + j];
    W2[j * 175 + i] = Wm[j * 175 + i];  // benign same-value races
  }
}

__global__ void k_degree(const float* W2, float* dsi) {
  int j = threadIdx.x;
  if (j >= 175) return;
  float s = 0.f;
  for (int i = 0; i < 175; ++i) s += W2[i * 175 + j];
  dsi[j] = sqrtf(1.f / (s + EPSF));
}

// Aug = [ I - alpha*S + EPS | y ], 175 x 180
__global__ void k_build_aug(const float* W2, const float* dsi, const float* alpha_p,
                            float* Aug) {
  int idx = blockIdx.x * blockDim.x + threadIdx.x;
  if (idx >= 175 * 180) return;
  int i = idx / 180, j = idx - 180 * i;
  float alpha = alpha_p[0];
  if (j < 175) {
    Aug[idx] = ((i == j) ? 1.f : 0.f) - alpha * dsi[i] * W2[i * 175 + j] * dsi[j] + EPSF;
  } else {
    int c = j - 175;
    Aug[idx] = (i < 25) ? ((c == i / 5) ? 1.f : 0.f) : EPSF;
  }
}

// Gauss-Jordan with partial pivoting; solves M F = y for the 5 RHS columns.
__global__ void k_gauss_solve(float* A, float* F) {
  __shared__ int s_piv;
  __shared__ float s_inv;
  const int tid = threadIdx.x;
  for (int k = 0; k < 175; ++k) {
    if (tid == 0) {
      int p = k; float best = fabsf(A[k * 180 + k]);
      for (int i = k + 1; i < 175; ++i) {
        float v = fabsf(A[i * 180 + k]);
        if (v > best) { best = v; p = i; }
      }
      s_piv = p;
    }
    __syncthreads();
    int p = s_piv;
    if (p != k) {
      for (int j = tid; j < 180; j += 256) {
        float t = A[k * 180 + j]; A[k * 180 + j] = A[p * 180 + j]; A[p * 180 + j] = t;
      }
    }
    __syncthreads();
    if (tid == 0) s_inv = 1.f / A[k * 180 + k];
    __syncthreads();
    float inv = s_inv;
    for (int i = tid; i < 175; i += 256) {
      if (i == k) continue;
      float f = A[i * 180 + k] * inv;
      if (f != 0.f)
        for (int j = k; j < 180; ++j) A[i * 180 + j] -= f * A[k * 180 + j];
    }
    __syncthreads();
  }
  for (int i = tid; i < 175; i += 256) {
    float inv = 1.f / A[i * 180 + i];
    for (int c = 0; c < 5; ++c) F[i * 5 + c] = A[i * 180 + 175 + c] * inv;
  }
}

// ------------------------------------------------------------- prototypical loss
__global__ void k_proto_mean(const float* feat, float* zp) {
  int i = blockIdx.x * blockDim.x + threadIdx.x;
  if (i >= 5 * 1600) return;
  int c = i / 1600, d = i - c * 1600;
  float s = 0.f;
  for (int k = 0; k < 5; ++k) s += feat[(c * 5 + k) * 1600 + d];
  zp[i] = s * 0.2f;
}

__global__ void k_proto_logp(const float* feat, const float* zp, float* logpq) {
  int qi = blockIdx.x, tid = threadIdx.x;
  const float* qf = feat + (size_t)(25 + qi) * 1600;
  float dot[5] = {0, 0, 0, 0, 0}, pp[5] = {0, 0, 0, 0, 0}, qq = 0.f;
  for (int d = tid; d < 1600; d += 64) {
    float qv = qf[d]; qq += qv * qv;
    for (int c = 0; c < 5; ++c) {
      float pv = zp[c * 1600 + d];
      dot[c] += qv * pv; pp[c] += pv * pv;
    }
  }
  __shared__ float red[64 * 11];
  red[tid * 11 + 0] = qq;
  for (int c = 0; c < 5; ++c) { red[tid * 11 + 1 + c] = dot[c]; red[tid * 11 + 6 + c] = pp[c]; }
  __syncthreads();
  if (tid == 0) {
    float acc[11];
    for (int k = 0; k < 11; ++k) acc[k] = 0.f;
    for (int t = 0; t < 64; ++t)
      for (int k = 0; k < 11; ++k) acc[k] += red[t * 11 + k];
    float d5[5], m = -1e30f;
    for (int c = 0; c < 5; ++c) {
      d5[c] = fmaxf(acc[0] + acc[6 + c] - 2.f * acc[1 + c], 0.f) * (1.f / 1600.f);
      float lg = -d5[c];
      if (lg > m) m = lg;
    }
    float s = 0.f;
    for (int c = 0; c < 5; ++c) s += expf(-d5[c] - m);
    int tcls = qi / 30;
    logpq[qi] = (-d5[tcls] - m) - logf(s);
  }
}

// ------------------------------------------------------------- final reductions
__global__ void k_finalize(const float* F, const float* logpq, float* out) {
  __shared__ float s_ce[256], s_ac[256], s_pl[256];
  int tid = threadIdx.x;
  float ce = 0.f, ac = 0.f, pl = 0.f;
  for (int i = tid; i < 175; i += 256) {
    float f[5], m = -1e30f;
    for (int c = 0; c < 5; ++c) { f[c] = F[i * 5 + c]; if (f[c] > m) m = f[c]; }
    float e[5], s = 0.f;
    for (int c = 0; c < 5; ++c) { e[c] = expf(f[c] - m); s += e[c]; }
    int t = (i < 25) ? (i / 5) : ((i - 25) / 30);
    ce += -logf(e[t] / s + EPSF);
    if (i >= 25) {
      int lbl = 0; float bv = f[0];
      for (int c = 1; c < 5; ++c) if (f[c] > bv) { bv = f[c]; lbl = c; }
      if (lbl == t) ac += 1.f;
    }
  }
  for (int qi = tid; qi < 150; qi += 256) pl += logpq[qi];
  s_ce[tid] = ce; s_ac[tid] = ac; s_pl[tid] = pl;
  __syncthreads();
  if (tid == 0) {
    float a = 0.f, b = 0.f, c = 0.f;
    for (int t = 0; t < 256; ++t) { a += s_ce[t]; b += s_ac[t]; c += s_pl[t]; }
    out[0] = -c / 150.f;   // proto_loss
    out[1] = a / 175.f;    // ce_loss
    out[2] = b / 150.f;    // acc
  }
}

// ================================================================= host launch
#define GRID(n) dim3(((n) + 255) / 256)

extern "C" void kernel_launch(void* const* d_in, const int* in_sizes, int n_in,
                              void* d_out, int out_size, void* d_ws, size_t ws_size,
                              hipStream_t stream) {
  (void)out_size; (void)ws_size; (void)n_in;
  // Input index maps. Primary: jax-pytree sorted-key flattening
  //   [s, q, alpha, enc0(w,b,g,be,mm,mv)..enc3, rel.b1, rel.b2, rel.c1(6), rel.c2(6), w1, w2]
  // Fallback: insertion order [s, q, enc.., rel.c1, rel.c2, w1, b1, w2, b2, alpha]
  int ia, encBase, ic1, ic2, iw1, ib1, iw2, ib2;
  if (in_sizes[2] == 1) {
    ia = 2; encBase = 3; ib1 = 27; ib2 = 28; ic1 = 29; ic2 = 35; iw1 = 41; iw2 = 42;
  } else {
    encBase = 2; ic1 = 26; ic2 = 32; iw1 = 38; ib1 = 39; iw2 = 40; ib2 = 41; ia = 42;
  }
  const float* S = (const float*)d_in[0];
  const float* Q = (const float*)d_in[1];
  const float* alpha = (const float*)d_in[ia];
  float* OUT = (float*)d_out;

  auto P = [&](int i) { return (const float*)d_in[i]; };

  // ---- workspace carving
  char* base = (char*)d_ws;
  size_t off = 0;
  auto alloc = [&](size_t bytes) -> void* {
    off = (off + 255) & ~(size_t)255;
    void* p = base + off;
    off += bytes;
    return p;
  };
  _Float16* wts1h = (_Float16*)alloc((size_t)32 * 64 * 2);
  _Float16* wtsh[3];
  for (int l = 0; l < 3; ++l) wtsh[l] = (_Float16*)alloc((size_t)576 * 64 * 2);
  _Float16* wc1h = (_Float16*)alloc((size_t)576 * 64 * 2);
  float *sc[6], *bi[6];
  for (int l = 0; l < 5; ++l) { sc[l] = (float*)alloc(64 * 4); bi[l] = (float*)alloc(64 * 4); }
  sc[5] = (float*)alloc(4); bi[5] = (float*)alloc(4);
  _Float16* bufIn = (_Float16*)alloc((size_t)25 * 84 * 84 * 3 * 2);
  _Float16* bufA  = (_Float16*)alloc((size_t)25 * 84 * 84 * 64 * 2);
  _Float16* bufB  = (_Float16*)alloc((size_t)25 * 42 * 42 * 64 * 2);
  float*    feat  = (float*)alloc((size_t)175 * 1600 * 4);
  _Float16* featH = (_Float16*)alloc((size_t)175 * 1600 * 2);
  _Float16* r1    = (_Float16*)alloc((size_t)175 * 25 * 64 * 2);
  _Float16* r1p   = (_Float16*)alloc((size_t)175 * 9 * 64 * 2);
  float* sigma = (float*)alloc(175 * 4);
  float* xnp   = (float*)alloc((size_t)176 * 1600 * 4);
  float* G     = (float*)alloc((size_t)176 * 176 * 4);
  float* Wm    = (float*)alloc((size_t)175 * 175 * 4);
  float* W2    = (float*)alloc((size_t)175 * 175 * 4);
  float* dsi   = (float*)alloc(175 * 4);
  float* Aug   = (float*)alloc((size_t)175 * 180 * 4);
  float* Fm    = (float*)alloc((size_t)175 * 5 * 4);
  float* zp    = (float*)alloc((size_t)5 * 1600 * 4);
  float* logpq = (float*)alloc(150 * 4);

  // ---- per-call prep
  k_cvt_wsw<<<GRID(32 * 64), 256, 0, stream>>>(P(encBase + 0), wts1h, 27, 32);
  for (int l = 1; l < 4; ++l)
    k_cvt_wsw<<<GRID(576 * 64), 256, 0, stream>>>(P(encBase + 6 * l), wtsh[l - 1], 576, 576);
  k_cvt_wsw<<<GRID(576 * 64), 256, 0, stream>>>(P(ic1 + 0), wc1h, 576, 576);
  for (int l = 0; l < 4; ++l) {
    int b0 = encBase + 6 * l;
    k_bnprep<<<1, 64, 0, stream>>>(P(b0 + 2), P(b0 + 3), P(b0 + 4), P(b0 + 5), P(b0 + 1),
                                   sc[l], bi[l], 64);
  }
  k_bnprep<<<1, 64, 0, stream>>>(P(ic1 + 2), P(ic1 + 3), P(ic1 + 4), P(ic1 + 5), P(ic1 + 1),
                                 sc[4], bi[4], 64);
  k_bnprep<<<1, 64, 0, stream>>>(P(ic2 + 2), P(ic2 + 3), P(ic2 + 4), P(ic2 + 5), P(ic2 + 1),
                                 sc[5], bi[5], 1);
  k_fill_zero<<<GRID(176 * 1600), 256, 0, stream>>>(xnp, 176 * 1600);
  k_fill_zero<<<GRID(175 * 175), 256, 0, stream>>>(W2, 175 * 175);

  // ---- encoder: 7 chunks of 25 images (support first, then queries)
  for (int chunk = 0; chunk < 7; ++chunk) {
    int b0 = chunk * 25;
    k_pack_input<<<GRID(25 * 84 * 84 * 3), 256, 0, stream>>>(S, Q, bufIn, b0);
    k_conv3x3_wmma<<<dim3(441, 25), 128, 0, stream>>>(bufIn, wts1h, sc[0], bi[0], bufA,
                                                      84, 84, 3, 32);
    k_maxpool2<<<GRID(25 * 42 * 42 * 64), 256, 0, stream>>>(bufA, bufB, nullptr, 25, 84, 84, 42, 42);
    k_conv3x3_wmma<<<dim3(111, 25), 128, 0, stream>>>(bufB, wtsh[0], sc[1], bi[1], bufA,
                                                      42, 42, 64, 576);
    k_maxpool2<<<GRID(25 * 21 * 21 * 64), 256, 0, stream>>>(bufA, bufB, nullptr, 25, 42, 42, 21, 21);
    k_conv3x3_wmma<<<dim3(28, 25), 128, 0, stream>>>(bufB, wtsh[1], sc[2], bi[2], bufA,
                                                     21, 21, 64, 576);
    k_maxpool2<<<GRID(25 * 10 * 10 * 64), 256, 0, stream>>>(bufA, bufB, nullptr, 25, 21, 21, 10, 10);
    k_conv3x3_wmma<<<dim3(7, 25), 128, 0, stream>>>(bufB, wtsh[2], sc[3], bi[3], bufA,
                                                    10, 10, 64, 576);
    k_maxpool2<<<GRID(25 * 5 * 5 * 64), 256, 0, stream>>>(bufA, nullptr,
                                                          feat + (size_t)b0 * 1600,
                                                          25, 10, 10, 5, 5);
  }

  // ---- relation net -> sigma
  k_f32_to_f16<<<GRID(175 * 1600), 256, 0, stream>>>(feat, featH, 175 * 1600);
  k_conv3x3_wmma<<<dim3(2, 175), 128, 0, stream>>>(featH, wc1h, sc[4], bi[4], r1,
                                                   5, 5, 64, 576);
  k_maxpool2<<<GRID(175 * 3 * 3 * 64), 256, 0, stream>>>(r1, r1p, nullptr, 175, 5, 5, 3, 3);
  k_relation_tail<<<175, 32, 0, stream>>>(r1p, P(ic2 + 0), sc[5], bi[5],
                                          P(iw1), P(ib1), P(iw2), P(ib2), sigma);

  // ---- graph + label propagation
  k_build_xn<<<GRID(175 * 1600), 256, 0, stream>>>(feat, sigma, xnp);
  k_gram_wmma<<<31, 128, 0, stream>>>(xnp, G);
  k_wm<<<GRID(175 * 175), 256, 0, stream>>>(G, Wm);
  k_topk_mask<<<1, 256, 0, stream>>>(Wm, W2);
  k_degree<<<1, 256, 0, stream>>>(W2, dsi);
  k_build_aug<<<GRID(175 * 180), 256, 0, stream>>>(W2, dsi, alpha, Aug);
  k_gauss_solve<<<1, 256, 0, stream>>>(Aug, Fm);

  // ---- prototypical loss
  k_proto_mean<<<GRID(5 * 1600), 256, 0, stream>>>(feat, zp);
  k_proto_logp<<<150, 64, 0, stream>>>(feat, zp, logpq);

  // ---- outputs
  k_finalize<<<1, 256, 0, stream>>>(Fm, logpq, OUT);
}
